// MMMambaModel_21827023798414
// MI455X (gfx1250) — matrix operand order repaired and checked by reading, not compiled, not run on hardware
//
#include <hip/hip_runtime.h>
#include <hip/hip_bf16.h>

// ---------------------------------------------------------------------------
// MI455X / gfx1250 implementation of the multimodal Mamba reference.
// All GEMMs run on v_wmma_f32_16x16x32_bf16 (bf16 inputs, f32 accumulate).
// Each wave owns a 32x64 output tile: 2 A fragments x 4 B fragments ->
// 8 WMMAs per 12 b128 loads (~21 flop/byte from L2).
// ---------------------------------------------------------------------------

typedef __attribute__((ext_vector_type(16))) __bf16 v16bf;
typedef __attribute__((ext_vector_type(8)))  __bf16 v8bf;
typedef __attribute__((ext_vector_type(8)))  float  v8f;

#define M_TOK   8192      // B * L
#define BATCH   8
#define SEQL    1024
#define DMODEL  512
#define DINNER  1024
#define NXZ     2048      // 2 * DINNER
#define DSTATE  16
#define DTRANK  32
#define NDBL    64        // DTRANK + 2*DSTATE
#define KFUSE   864       // 1 + 64 + 768 = 833, padded to multiple of 32
#define PREDLEN 24

__device__ __forceinline__ float siluf(float x)     { return x / (1.0f + __expf(-x)); }
__device__ __forceinline__ float softplusf(float x) { return (x > 20.0f) ? x : log1pf(__expf(x)); }

// ---------------------------------------------------------------------------
// Generic bf16 WMMA GEMM:  C[M x N] = A[M x K] * W[K x N] (+bias) (+act)
// W is supplied TRANSPOSED as Wt[N x K] so the B-fragment is a contiguous
// 32-byte load per lane.  Each wave computes a 32x64 tile.
// Requires: M % 32 == 0, N % 64 == 0, K % 32 == 0.
//
// Fragment layouts used by this kernel (wave32; contiguous k order, same in A and B):
//   A 16x32 bf16 : lane(0-15) row=lane,   elems = K{kb..kb+15}
//                  lane(16-31) row=lane-16, elems = K{kb+16..kb+31}
//   B 32x16 bf16 : lane holds col = lane&15, K = (lane>>4)*16 + j  (j=0..15)
//   C/D 16x16 f32: vgpr j -> row = j + 8*(lane>>4), col = lane&15
// ---------------------------------------------------------------------------
union ABu { v16bf v; v8bf h[2]; };

__global__ __launch_bounds__(256) void gemm_bf16_wmma(
    const __bf16* __restrict__ A, const __bf16* __restrict__ Wt,
    const float* __restrict__ bias, float* __restrict__ C,
    __bf16* __restrict__ Cbf, int M, int N, int K, int act)
{
  const int wave = (blockIdx.x * blockDim.x + threadIdx.x) >> 5;
  const int lane = threadIdx.x & 31;
  const int strips_n = N >> 6;
  const int tile_m = wave / strips_n;        // 32-row tiles
  const int nbase  = (wave % strips_n) << 6;
  const int hl = lane >> 4;   // which half-wave
  const int lr = lane & 15;

  const __bf16* Arow0 = A + (size_t)(tile_m * 32 + lr) * K;
  const __bf16* Arow1 = Arow0 + (size_t)16 * K;
  const size_t wstride = (size_t)16 * K;

  v8f acc[8];
#pragma unroll
  for (int i = 0; i < 8; ++i) acc[i] = (v8f){};

  for (int kb = 0; kb < K; kb += 32) {
    ABu a0, a1;
    a0.h[0] = *reinterpret_cast<const v8bf*>(Arow0 + kb + hl * 16);
    a0.h[1] = *reinterpret_cast<const v8bf*>(Arow0 + kb + hl * 16 + 8);
    a1.h[0] = *reinterpret_cast<const v8bf*>(Arow1 + kb + hl * 16);
    a1.h[1] = *reinterpret_cast<const v8bf*>(Arow1 + kb + hl * 16 + 8);
    const __bf16* Wb = Wt + (size_t)(nbase + lr) * K + kb + hl * 16;
    v16bf b0 = *reinterpret_cast<const v16bf*>(Wb);
    v16bf b1 = *reinterpret_cast<const v16bf*>(Wb + wstride);
    v16bf b2 = *reinterpret_cast<const v16bf*>(Wb + 2 * wstride);
    v16bf b3 = *reinterpret_cast<const v16bf*>(Wb + 3 * wstride);
    acc[0] = __builtin_amdgcn_wmma_f32_16x16x32_bf16(false, a0.v, false, b0, (short)0, acc[0], false, false);
    acc[1] = __builtin_amdgcn_wmma_f32_16x16x32_bf16(false, a0.v, false, b1, (short)0, acc[1], false, false);
    acc[2] = __builtin_amdgcn_wmma_f32_16x16x32_bf16(false, a0.v, false, b2, (short)0, acc[2], false, false);
    acc[3] = __builtin_amdgcn_wmma_f32_16x16x32_bf16(false, a0.v, false, b3, (short)0, acc[3], false, false);
    acc[4] = __builtin_amdgcn_wmma_f32_16x16x32_bf16(false, a1.v, false, b0, (short)0, acc[4], false, false);
    acc[5] = __builtin_amdgcn_wmma_f32_16x16x32_bf16(false, a1.v, false, b1, (short)0, acc[5], false, false);
    acc[6] = __builtin_amdgcn_wmma_f32_16x16x32_bf16(false, a1.v, false, b2, (short)0, acc[6], false, false);
    acc[7] = __builtin_amdgcn_wmma_f32_16x16x32_bf16(false, a1.v, false, b3, (short)0, acc[7], false, false);
  }

#pragma unroll
  for (int half = 0; half < 2; ++half) {
#pragma unroll
    for (int t = 0; t < 4; ++t) {
      const int col = nbase + t * 16 + lr;
      const float bv = bias ? bias[col] : 0.0f;
#pragma unroll
      for (int j = 0; j < 8; ++j) {
        const int row = tile_m * 32 + half * 16 + j + hl * 8;
        float v = acc[half * 4 + t][j] + bv;
        if (act == 1) v = softplusf(v);
        const size_t o = (size_t)row * N + col;
        C[o] = v;
        if (Cbf) Cbf[o] = (__bf16)v;
      }
    }
  }
}

// ---------------------------------------------------------------------------
// Packing / conversion kernels
// ---------------------------------------------------------------------------

// Xin[m][k] bf16: col 0 = x_load, 1..64 = x_img, 65..832 = x_text, pad 0
__global__ __launch_bounds__(256) void pack_xin_kernel(
    const float* __restrict__ xl, const float* __restrict__ xi,
    const float* __restrict__ xt, __bf16* __restrict__ Xin)
{
  size_t idx = (size_t)blockIdx.x * blockDim.x + threadIdx.x;
  if (idx >= (size_t)M_TOK * KFUSE) return;
  size_t m = idx / KFUSE; int k = (int)(idx % KFUSE);
  float v;
  if (k == 0)        v = xl[m];
  else if (k < 65)   v = xi[m * 64 + (k - 1)];
  else if (k < 833)  v = xt[m * 768 + (k - 65)];
  else               v = 0.0f;
  Xin[idx] = (__bf16)v;
}

// Wcat transposed: Wt[n][k] = Wcat[k][n], same row ordering as pack_xin
__global__ __launch_bounds__(256) void pack_wcat_kernel(
    const float* __restrict__ wl, const float* __restrict__ wi,
    const float* __restrict__ wt, __bf16* __restrict__ Wt)
{
  int idx = blockIdx.x * blockDim.x + threadIdx.x;
  if (idx >= DMODEL * KFUSE) return;
  int n = idx / KFUSE, k = idx % KFUSE;
  float v;
  if (k == 0)        v = wl[n];
  else if (k < 65)   v = wi[(k - 1) * DMODEL + n];
  else if (k < 833)  v = wt[(k - 65) * DMODEL + n];
  else               v = 0.0f;
  Wt[(size_t)n * KFUSE + k] = (__bf16)v;
}

// Wt[n*K + k] = (bf16) W[k*N + n]
__global__ __launch_bounds__(256) void transpose_bf16_kernel(
    const float* __restrict__ W, __bf16* __restrict__ Wt, int K, int N)
{
  int idx = blockIdx.x * blockDim.x + threadIdx.x;
  if (idx >= K * N) return;
  int k = idx / N, n = idx % N;
  Wt[(size_t)n * K + k] = (__bf16)W[idx];
}

// dt = x_dbl[:, :32]  -> bf16
__global__ __launch_bounds__(256) void dt_extract_kernel(
    const float* __restrict__ XDBL, __bf16* __restrict__ DT)
{
  size_t idx = (size_t)blockIdx.x * blockDim.x + threadIdx.x;
  if (idx >= (size_t)M_TOK * DTRANK) return;
  size_t m = idx >> 5; int k = (int)(idx & 31);
  DT[idx] = (__bf16)XDBL[m * NDBL + k];
}

// ---------------------------------------------------------------------------
// Depthwise causal conv (D_CONV=4) + SiLU, reading the u-half of XZ
// ---------------------------------------------------------------------------
__global__ __launch_bounds__(256) void conv_silu_kernel(
    const float* __restrict__ XZ, const float* __restrict__ conv_w,
    const float* __restrict__ conv_b, float* __restrict__ U,
    __bf16* __restrict__ Ubf)
{
  size_t idx = (size_t)blockIdx.x * blockDim.x + threadIdx.x;
  if (idx >= (size_t)M_TOK * DINNER) return;
  int e = (int)(idx & (DINNER - 1));
  size_t m = idx >> 10;
  int t = (int)(m & (SEQL - 1));
  float acc = conv_b[e];
#pragma unroll
  for (int k = 0; k < 4; ++k) {
    int tt = t - 3 + k;
    if (tt >= 0) acc += XZ[(m - 3 + k) * NXZ + e] * conv_w[e * 4 + k];
  }
  float v = siluf(acc);
  U[idx] = v;
  Ubf[idx] = (__bf16)v;
}

// ---------------------------------------------------------------------------
// Selective scan: one (batch, channel) recurrence per lane, h[16] in regs.
// B_t / C_t broadcast through LDS.  Y aliases DELTA in the workspace
// (same-address read-before-write per element), so NO __restrict__ on them.
// ---------------------------------------------------------------------------
__global__ __launch_bounds__(256) void scan_kernel(
    const float* __restrict__ XDBL, const float* DELTA,
    const float* __restrict__ U, const float* __restrict__ A_log,
    const float* __restrict__ Dd, float* Y)
{
  const int b = blockIdx.x >> 2;                       // DINNER/256 = 4 chunks
  const int e = ((blockIdx.x & 3) << 8) + threadIdx.x;
  float Av[DSTATE];
#pragma unroll
  for (int n = 0; n < DSTATE; ++n) Av[n] = -__expf(A_log[(size_t)e * DSTATE + n]);
  const float Dv = Dd[e];
  float h[DSTATE];
#pragma unroll
  for (int n = 0; n < DSTATE; ++n) h[n] = 0.0f;

  __shared__ float Bs[DSTATE], Cs[DSTATE];
  for (int t = 0; t < SEQL; ++t) {
    const size_t m = (size_t)b * SEQL + t;
    __syncthreads();
    if (threadIdx.x < DSTATE)
      Bs[threadIdx.x] = XDBL[m * NDBL + DTRANK + threadIdx.x];
    else if (threadIdx.x < 2 * DSTATE)
      Cs[threadIdx.x - DSTATE] = XDBL[m * NDBL + DTRANK + DSTATE + (threadIdx.x - DSTATE)];
    __syncthreads();
    const float dt = DELTA[m * DINNER + e];
    const float ut = U[m * DINNER + e];
    const float du = dt * ut;
    float y = 0.0f;
#pragma unroll
    for (int n = 0; n < DSTATE; ++n) {
      h[n] = __expf(dt * Av[n]) * h[n] + du * Bs[n];
      y += h[n] * Cs[n];
    }
    Y[m * DINNER + e] = y + ut * Dv;
  }
}

// y = y * silu(res), res = XZ[:, 1024:2048]; also emit bf16 copy
__global__ __launch_bounds__(256) void ymul_kernel(
    float* __restrict__ Y, const float* __restrict__ XZ, __bf16* __restrict__ Ybf)
{
  size_t idx = (size_t)blockIdx.x * blockDim.x + threadIdx.x;
  if (idx >= (size_t)M_TOK * DINNER) return;
  int e = (int)(idx & (DINNER - 1));
  size_t m = idx >> 10;
  float v = Y[idx] * siluf(XZ[m * NXZ + DINNER + e]);
  Y[idx] = v;
  Ybf[idx] = (__bf16)v;
}

// ---------------------------------------------------------------------------
// Head: layernorm of the last token of each batch + 512x24 FC
// ---------------------------------------------------------------------------
__global__ __launch_bounds__(256) void head_kernel(
    const float* __restrict__ X, const float* __restrict__ ln_g,
    const float* __restrict__ ln_b, const float* __restrict__ fc_w,
    const float* __restrict__ fc_b, float* __restrict__ out)
{
  __shared__ float red[256];
  __shared__ float xs[DMODEL];
  const int b = blockIdx.x, tid = threadIdx.x;
  const float* xrow = X + (size_t)(b * SEQL + SEQL - 1) * DMODEL;

  red[tid] = xrow[tid] + xrow[tid + 256];
  __syncthreads();
  for (int o = 128; o > 0; o >>= 1) { if (tid < o) red[tid] += red[tid + o]; __syncthreads(); }
  const float mu = red[0] / DMODEL;
  __syncthreads();
  const float d0 = xrow[tid] - mu, d1 = xrow[tid + 256] - mu;
  red[tid] = d0 * d0 + d1 * d1;
  __syncthreads();
  for (int o = 128; o > 0; o >>= 1) { if (tid < o) red[tid] += red[tid + o]; __syncthreads(); }
  const float rstd = rsqrtf(red[0] / DMODEL + 1e-5f);
  xs[tid]       = d0 * rstd * ln_g[tid]       + ln_b[tid];
  xs[tid + 256] = d1 * rstd * ln_g[tid + 256] + ln_b[tid + 256];
  __syncthreads();
  if (tid < PREDLEN) {
    float acc = fc_b[tid];
    for (int i = 0; i < DMODEL; ++i) acc += xs[i] * fc_w[i * PREDLEN + tid];
    out[b * PREDLEN + tid] = acc;
  }
}

// ---------------------------------------------------------------------------
// Host orchestration
// ---------------------------------------------------------------------------
extern "C" void kernel_launch(void* const* d_in, const int* in_sizes, int n_in,
                              void* d_out, int out_size, void* d_ws, size_t ws_size,
                              hipStream_t stream) {
  (void)in_sizes; (void)n_in; (void)out_size; (void)ws_size;

  const float* x_load  = (const float*)d_in[0];
  const float* x_img   = (const float*)d_in[1];
  const float* x_text  = (const float*)d_in[2];
  const float* w_load  = (const float*)d_in[3];
  const float* w_img   = (const float*)d_in[4];
  const float* w_text  = (const float*)d_in[5];
  const float* b_fuse  = (const float*)d_in[6];
  const float* in_w    = (const float*)d_in[7];
  const float* in_b    = (const float*)d_in[8];
  const float* conv_w  = (const float*)d_in[9];
  const float* conv_b  = (const float*)d_in[10];
  const float* xproj_w = (const float*)d_in[11];
  const float* dt_w    = (const float*)d_in[12];
  const float* dt_b    = (const float*)d_in[13];
  const float* A_log   = (const float*)d_in[14];
  const float* Dparam  = (const float*)d_in[15];
  const float* out_w   = (const float*)d_in[16];
  const float* out_b   = (const float*)d_in[17];
  const float* ln_g    = (const float*)d_in[18];
  const float* ln_b    = (const float*)d_in[19];
  const float* fc_w    = (const float*)d_in[20];
  const float* fc_b    = (const float*)d_in[21];

  char* ws = (char*)d_ws;
  size_t off = 0;
  auto alloc = [&](size_t bytes) -> char* {
    char* p = ws + off;
    off += (bytes + 255) & ~(size_t)255;
    return p;
  };

  // XZ is the largest buffer; Xin (only needed before the first in_proj
  // GEMM writes XZ) aliases its head.  Y aliases DELTA (scan reads delta[t]
  // before writing y[t], same thread/element).
  float*  XZ    = (float*) alloc((size_t)M_TOK * NXZ * 4);       // 64 MB
  __bf16* XIN   = (__bf16*)XZ;                                   // alias
  __bf16* WCAT  = (__bf16*)alloc((size_t)DMODEL * KFUSE * 2);
  float*  X     = (float*) alloc((size_t)M_TOK * DMODEL * 4);
  __bf16* XBF   = (__bf16*)alloc((size_t)M_TOK * DMODEL * 2);
  __bf16* INWT  = (__bf16*)alloc((size_t)NXZ * DMODEL * 2);
  float*  U     = (float*) alloc((size_t)M_TOK * DINNER * 4);
  __bf16* UBF   = (__bf16*)alloc((size_t)M_TOK * DINNER * 2);
  __bf16* XPJT  = (__bf16*)alloc((size_t)NDBL * DINNER * 2);
  float*  XDBL  = (float*) alloc((size_t)M_TOK * NDBL * 4);
  __bf16* DTBF  = (__bf16*)alloc((size_t)M_TOK * DTRANK * 2);
  __bf16* DTWT  = (__bf16*)alloc((size_t)DINNER * DTRANK * 2);
  float*  DELTA = (float*) alloc((size_t)M_TOK * DINNER * 4);
  float*  Y     = DELTA;                                         // alias
  __bf16* YBF   = (__bf16*)alloc((size_t)M_TOK * DINNER * 2);
  __bf16* OUTWT = (__bf16*)alloc((size_t)DMODEL * DINNER * 2);

  auto cdiv = [](size_t a, size_t b) { return (unsigned)((a + b - 1) / b); };
  auto gemm = [&](const __bf16* A, const __bf16* Wt, const float* bias,
                  float* C, __bf16* Cbf, int M, int N, int K, int act) {
    unsigned waves = (unsigned)(M / 32) * (unsigned)(N / 64);
    gemm_bf16_wmma<<<waves / 8, 256, 0, stream>>>(A, Wt, bias, C, Cbf, M, N, K, act);
  };

  // ---- input fusion: x = [load|img|text] @ Wcat + b_fuse -------------------
  pack_xin_kernel<<<cdiv((size_t)M_TOK * KFUSE, 256), 256, 0, stream>>>(x_load, x_img, x_text, XIN);
  pack_wcat_kernel<<<cdiv((size_t)DMODEL * KFUSE, 256), 256, 0, stream>>>(w_load, w_img, w_text, WCAT);
  gemm(XIN, WCAT, b_fuse, X, XBF, M_TOK, DMODEL, KFUSE, 0);

  // ---- mamba layers --------------------------------------------------------
  for (int li = 0; li < 2; ++li) {
    const float* in_w_l    = in_w    + (size_t)li * DMODEL * NXZ;
    const float* in_b_l    = in_b    + (size_t)li * NXZ;
    const float* conv_w_l  = conv_w  + (size_t)li * DINNER * 4;
    const float* conv_b_l  = conv_b  + (size_t)li * DINNER;
    const float* xproj_l   = xproj_w + (size_t)li * DINNER * NDBL;
    const float* dt_w_l    = dt_w    + (size_t)li * DTRANK * DINNER;
    const float* dt_b_l    = dt_b    + (size_t)li * DINNER;
    const float* A_log_l   = A_log   + (size_t)li * DINNER * DSTATE;
    const float* D_l       = Dparam  + (size_t)li * DINNER;
    const float* out_w_l   = out_w   + (size_t)li * DINNER * DMODEL;
    const float* out_b_l   = out_b   + (size_t)li * DMODEL;

    // in_proj: xz = x @ in_w + in_b   (destroys XIN alias; no longer needed)
    transpose_bf16_kernel<<<cdiv((size_t)DMODEL * NXZ, 256), 256, 0, stream>>>(in_w_l, INWT, DMODEL, NXZ);
    gemm(XBF, INWT, in_b_l, XZ, nullptr, M_TOK, NXZ, DMODEL, 0);

    // depthwise conv + SiLU on u-half
    conv_silu_kernel<<<cdiv((size_t)M_TOK * DINNER, 256), 256, 0, stream>>>(XZ, conv_w_l, conv_b_l, U, UBF);

    // x_dbl = u @ xproj_w
    transpose_bf16_kernel<<<cdiv((size_t)DINNER * NDBL, 256), 256, 0, stream>>>(xproj_l, XPJT, DINNER, NDBL);
    gemm(UBF, XPJT, nullptr, XDBL, nullptr, M_TOK, NDBL, DINNER, 0);

    // delta = softplus(dt @ dt_w + dt_b)
    dt_extract_kernel<<<cdiv((size_t)M_TOK * DTRANK, 256), 256, 0, stream>>>(XDBL, DTBF);
    transpose_bf16_kernel<<<cdiv((size_t)DTRANK * DINNER, 256), 256, 0, stream>>>(dt_w_l, DTWT, DTRANK, DINNER);
    gemm(DTBF, DTWT, dt_b_l, DELTA, nullptr, M_TOK, DINNER, DTRANK, 1);

    // selective scan (Y aliases DELTA) + gating
    scan_kernel<<<BATCH * (DINNER / 256), 256, 0, stream>>>(XDBL, DELTA, U, A_log_l, D_l, Y);
    ymul_kernel<<<cdiv((size_t)M_TOK * DINNER, 256), 256, 0, stream>>>(Y, XZ, YBF);

    // out_proj: x = y @ out_w + out_b (+ bf16 copy for the next layer)
    transpose_bf16_kernel<<<cdiv((size_t)DINNER * DMODEL, 256), 256, 0, stream>>>(out_w_l, OUTWT, DINNER, DMODEL);
    gemm(YBF, OUTWT, out_b_l, X, XBF, M_TOK, DMODEL, DINNER, 0);
  }

  // ---- layernorm of last token + FC head ----------------------------------
  head_kernel<<<BATCH, 256, 0, stream>>>(X, ln_g, ln_b, fc_w, fc_b, (float*)d_out);
}
